// RelationGraphSAGE_6485400617280
// MI455X (gfx1250) — matrix operand order repaired and checked by reading, not compiled, not run on hardware
//
#include <hip/hip_runtime.h>
#include <hip/hip_bf16.h>

// ---------------------------------------------------------------------------
// RelationGraphSAGE forward, CDNA5 (gfx1250), wave32.
//   Stage 1 (agg1 = mask1 @ feats, 536 MB HBM-bound): TDM (tensor_load_to_lds)
//   double-buffers the shared 8 KB B tile in LDS; A stream pipelined in regs;
//   V_WMMA_F32_16X16X32_F16 compute.
//   Stages 2-4: small GEMMs, register-resident fragments, batched B loads.
// ---------------------------------------------------------------------------

typedef __attribute__((ext_vector_type(16))) _Float16 v16h;
typedef __attribute__((ext_vector_type(8)))  _Float16 v8h;
typedef __attribute__((ext_vector_type(8)))  float    v8f;
typedef __attribute__((ext_vector_type(4)))  unsigned v4u;
typedef __attribute__((ext_vector_type(8)))  int      v8i;
typedef __attribute__((ext_vector_type(4)))  int      v4i;

#define M2   32768
#define M1   4096
#define M0   1024
#define DIN  128
#define DH   256
#define DOUT 256

__device__ __forceinline__ unsigned lds_off(const void* p) {
    // generic pointer to LDS: low 32 bits are the LDS byte address
    return (unsigned)(unsigned long long)p;
}

// ---------------------------------------------------------------------------
// Stage 1: agg1 = mask1 @ feats  (M=4096, K=32768, N=128)
//   - 8 waves/block, each wave owns a 16-row strip, all 8 N tiles
//   - B tile (featsT[0:128][kb:kb+32], 8 KB) staged by TDM into LDS, 2-deep
//   - A (mask1, f32) register-pipelined + converted to f16 in-flight
// ---------------------------------------------------------------------------
__global__ __launch_bounds__(256) void gemm_agg1_tdm(
    const float* __restrict__ mask1,      // [M1 x M2] f32
    const _Float16* __restrict__ featsT,  // [DIN x M2] f16 (row n = feature n)
    _Float16* __restrict__ Out, int ldo)  // agg1 -> X1[:,128:256]
{
    __shared__ __align__(64) _Float16 Bbuf[2][DIN * 32];   // 2 x 8 KB

    const int lane   = threadIdx.x & 31;
    const int wave   = threadIdx.x >> 5;
    const int mbase  = (blockIdx.x * 8 + wave) * 16;
    const int mrow   = mbase + (lane & 15);
    const int hihalf = lane >> 4;          // 0: lanes 0-15, 1: lanes 16-31
    const int aoff   = hihalf * 16;

    // ---- constant D# pieces (ISA ch.8): 2D tensor, data_size=2B ----
    v4i g2 = {0, 0, 0, 0};                 // groups 2/3 unused (2D tensor)
    v4i g3 = {0, 0, 0, 0};
    v8i g4 = {0, 0, 0, 0, 0, 0, 0, 0};     // aux group (6-arg clang-23 form)
    v8i g1;
    g1[0] = (1 << 16);                     // data_size code 1 (2 bytes), no mask
    g1[1] = (int)(0x8000u << 16);          // tensor_dim0 = 32768 (bits 79:48)
    g1[2] = (int)(128u << 16);             // tensor_dim1 = 128   (bits 111:80)
    g1[3] = (int)(32u  << 16);             // tile_dim0   = 32    (bits 127:112)
    g1[4] = 128;                           // tile_dim1   = 128, tile_dim2 = 0
    g1[5] = 32768;                         // tensor_dim0_stride = 32768 (lo32)
    g1[6] = 0;                             // stride hi16 = 0, dim1_stride = 0
    g1[7] = 0;

    const unsigned long long featsT_addr = (unsigned long long)featsT;
    auto issue_tdm = [&](int kb, int buf) {
        unsigned long long ga = featsT_addr + (unsigned long long)kb * 2ull;
        v4u g0;
        g0[0] = 1u;                                          // count = 1
        g0[1] = lds_off(&Bbuf[buf][0]);                      // lds_addr
        g0[2] = (unsigned)(ga & 0xFFFFFFFFu);                // global_addr lo
        g0[3] = (unsigned)((ga >> 32) & 0x1FFFFFFu) | (2u << 30); // hi | type=2
        __builtin_amdgcn_tensor_load_to_lds(g0, g1, g2, g3, g4, 0);
    };

    // prologue: first B tile
    if (wave == 0) {
        issue_tdm(0, 0);
        __builtin_amdgcn_s_wait_tensorcnt(0);
    }
    __syncthreads();

    const v8f zero = {};
    v8f acc[8];
#pragma unroll
    for (int t = 0; t < 8; ++t) acc[t] = zero;

    // prime A raw registers (16 f32 per lane per K-step)
    const float* ap = mask1 + (size_t)mrow * M2 + aoff;
    v8f alo = *(const v8f*)ap;
    v8f ahi = *(const v8f*)(ap + 8);

    for (int kb = 0; kb < M2; kb += 32) {
        const int  buf  = (kb >> 5) & 1;
        const bool more = (kb + 32) < M2;

        if (wave == 0 && more) issue_tdm(kb + 32, buf ^ 1);

        // start next A stream step before computing on the current one
        v8f nlo = zero, nhi = zero;
        if (more) {
            const float* pn = mask1 + (size_t)mrow * M2 + (kb + 32) + aoff;
            nlo = *(const v8f*)pn;
            nhi = *(const v8f*)(pn + 8);
            __builtin_prefetch(pn + 32, 0, 1);   // global_prefetch_b8
        }

        // A fragment (16x32 f16, contiguous k = 16*hihalf + s), converted f32->f16 in regs
        v16h afrag;
#pragma unroll
        for (int i = 0; i < 8; ++i) {
            afrag[i]     = (_Float16)alo[i];
            afrag[i + 8] = (_Float16)ahi[i];
        }

        // B fragments from LDS: column n, 16 contiguous halves
        v16h bf[8];
#pragma unroll
        for (int t = 0; t < 8; ++t) {
            const int n = t * 16 + (lane & 15);
            bf[t] = *(const v16h*)(&Bbuf[buf][n * 32 + hihalf * 16]);
        }
#pragma unroll
        for (int t = 0; t < 8; ++t)
            acc[t] = __builtin_amdgcn_wmma_f32_16x16x32_f16(
                false, afrag, false, bf[t], (short)0, acc[t], false, false);

        if (wave == 0 && more) __builtin_amdgcn_s_wait_tensorcnt(0);
        __syncthreads();     // publish next tile / retire current one

        alo = nlo; ahi = nhi;
    }

    // epilogue: f32 C layout -> half store
#pragma unroll
    for (int t = 0; t < 8; ++t) {
        const int n = t * 16 + (lane & 15);
#pragma unroll
        for (int i = 0; i < 8; ++i) {
            const int m = mbase + i + hihalf * 8;
            Out[(size_t)m * ldo + n] = (_Float16)acc[t][i];
        }
    }
}

// ---------------------------------------------------------------------------
// Generic wave32 WMMA GEMM:  D[M x (NT*16)] = A[M x K] * Bt[NT*16 x K]^T
// (used for the three small GEMMs). B loads batched in groups of 4 so the
// loadcnt waits amortize over several WMMAs.
// ---------------------------------------------------------------------------
template <int NT, bool A_F32, bool RELU, bool OUT_F32, bool WRITE_T>
__global__ __launch_bounds__(256) void gemm_wmma(
    const void* __restrict__ Asrc, int lda,
    const _Float16* __restrict__ Bt, int ldb,
    void* __restrict__ Out, int ldo,
    _Float16* __restrict__ OutT, int ldt,
    int K)
{
    const int lane   = threadIdx.x & 31;
    const int wave   = threadIdx.x >> 5;
    const int mbase  = (blockIdx.x * 8 + wave) * 16;
    const int mrow   = mbase + (lane & 15);
    const int hihalf = lane >> 4;
    const int aoff   = hihalf * 16;

    const v8f zero = {};
    v8f acc[NT];
#pragma unroll
    for (int t = 0; t < NT; ++t) acc[t] = zero;

    const float*    Af = (const float*)Asrc;
    const _Float16* Ah = (const _Float16*)Asrc;

    for (int kb = 0; kb < K; kb += 32) {
        v16h afrag;
        if (A_F32) {
            const float* p0 = Af + (size_t)mrow * lda + kb + aoff;
            v8f lo = *(const v8f*)p0;
            v8f hi = *(const v8f*)(p0 + 8);
#pragma unroll
            for (int i = 0; i < 8; ++i) {
                afrag[i]     = (_Float16)lo[i];
                afrag[i + 8] = (_Float16)hi[i];
            }
            if (kb + 32 < K) __builtin_prefetch(p0 + 32, 0, 1);
        } else {
            const _Float16* p0 = Ah + (size_t)mrow * lda + kb + aoff;
            v8h lo = *(const v8h*)p0;
            v8h hi = *(const v8h*)(p0 + 8);
#pragma unroll
            for (int i = 0; i < 8; ++i) {
                afrag[i]     = lo[i];
                afrag[i + 8] = hi[i];
            }
        }

#pragma unroll
        for (int g = 0; g < NT; g += 4) {
            v16h bf[4];
#pragma unroll
            for (int j = 0; j < 4; ++j) {
                const int n = (g + j) * 16 + (lane & 15);
                bf[j] = *(const v16h*)(Bt + (size_t)n * ldb + kb + hihalf * 16);
            }
#pragma unroll
            for (int j = 0; j < 4; ++j)
                acc[g + j] = __builtin_amdgcn_wmma_f32_16x16x32_f16(
                    false, afrag, false, bf[j], (short)0, acc[g + j], false, false);
        }
    }

#pragma unroll
    for (int t = 0; t < NT; ++t) {
        const int n = t * 16 + (lane & 15);
#pragma unroll
        for (int i = 0; i < 8; ++i) {
            const int m = mbase + i + hihalf * 8;
            float v = acc[t][i];
            if (RELU) v = v > 0.0f ? v : 0.0f;
            if (OUT_F32) ((float*)Out)[(size_t)m * ldo + n] = v;
            else         ((_Float16*)Out)[(size_t)m * ldo + n] = (_Float16)v;
            if (WRITE_T) OutT[(size_t)n * ldt + m] = (_Float16)v;
        }
    }
}

// ---------------------------------------------------------------------------
// Helpers
// ---------------------------------------------------------------------------
__global__ __launch_bounds__(256) void cvt_f32_f16(const float* __restrict__ s,
                                                   _Float16* __restrict__ d, int n)
{
    int i = blockIdx.x * 256 + threadIdx.x;
    if (i < n) d[i] = (_Float16)s[i];
}

__global__ __launch_bounds__(256) void transpose_feats(const float* __restrict__ feats,
                                                       _Float16* __restrict__ featsT)
{
    int i = blockIdx.x * 256 + threadIdx.x;      // over M2*DIN
    int node = i >> 7;
    int c    = i & 127;
    featsT[(size_t)c * M2 + node] = (_Float16)feats[i];
}

__global__ __launch_bounds__(256) void gather_self1(const float* __restrict__ feats,
                                                    const int* __restrict__ idx1,
                                                    _Float16* __restrict__ X1)
{
    int i = blockIdx.x * 256 + threadIdx.x;      // over M1*DIN
    int m = i >> 7;
    int c = i & 127;
    X1[(size_t)m * (2 * DIN) + c] = (_Float16)feats[(size_t)idx1[m] * DIN + c];
}

__global__ __launch_bounds__(256) void gather_self2(const _Float16* __restrict__ h1,
                                                    const int* __restrict__ idx0,
                                                    _Float16* __restrict__ X2)
{
    int i = blockIdx.x * 256 + threadIdx.x;      // over M0*DH
    int m = i >> 8;
    int c = i & 255;
    X2[(size_t)m * (2 * DH) + c] = h1[(size_t)idx0[m] * DH + c];
}

// ---------------------------------------------------------------------------
// Launch
// ---------------------------------------------------------------------------
extern "C" void kernel_launch(void* const* d_in, const int* in_sizes, int n_in,
                              void* d_out, int out_size, void* d_ws, size_t ws_size,
                              hipStream_t stream)
{
    const float* feats = (const float*)d_in[0];   // [32768 x 128]
    const float* mask1 = (const float*)d_in[1];   // [4096 x 32768]
    const float* mask0 = (const float*)d_in[2];   // [1024 x 4096]
    const float* W1    = (const float*)d_in[3];   // [256 x 256]
    const float* W2    = (const float*)d_in[4];   // [256 x 512]
    const int*   idx1  = (const int*)d_in[5];     // [4096]
    const int*   idx0  = (const int*)d_in[6];     // [1024]
    float*       out   = (float*)d_out;           // [1024 x 256]

    char* ws = (char*)d_ws;
    _Float16* featsT = (_Float16*)(ws);                       //  8 MB  [128 x 32768]
    _Float16* X1     = (_Float16*)(ws + 8388608);             //  2 MB  [4096 x 256]
    _Float16* h1     = (_Float16*)(ws + 10485760);            //  2 MB  [4096 x 256]
    _Float16* h1T    = (_Float16*)(ws + 12582912);            //  2 MB  [256 x 4096]
    _Float16* X2     = (_Float16*)(ws + 14680064);            //  1 MB  [1024 x 512]
    _Float16* W1h    = (_Float16*)(ws + 15728640);            // 128 KB [256 x 256]
    _Float16* W2h    = (_Float16*)(ws + 15859712);            // 256 KB [256 x 512]

    // stage 0: conversions / gathers
    cvt_f32_f16<<<(DH * 2 * DIN + 255) / 256, 256, 0, stream>>>(W1, W1h, DH * 2 * DIN);
    cvt_f32_f16<<<(DOUT * 2 * DH + 255) / 256, 256, 0, stream>>>(W2, W2h, DOUT * 2 * DH);
    transpose_feats<<<(M2 * DIN) / 256, 256, 0, stream>>>(feats, featsT);
    gather_self1<<<(M1 * DIN) / 256, 256, 0, stream>>>(feats, idx1, X1);

    // stage 1: agg1 = mask1 @ feats (TDM-staged B, WMMA f16) -> X1[:,128:256]
    gemm_agg1_tdm<<<M1 / 128, 256, 0, stream>>>(mask1, featsT, X1 + DIN, 2 * DIN);

    // stage 2: h1 = relu(X1 @ W1^T), also emit h1T
    gemm_wmma<16, false, true, false, true><<<M1 / 128, 256, 0, stream>>>(
        (const void*)X1, 2 * DIN, W1h, 2 * DIN,
        (void*)h1, DH, h1T, M1, 2 * DIN);

    gather_self2<<<(M0 * DH) / 256, 256, 0, stream>>>(h1, idx0, X2);

    // stage 3: agg0 = mask0 @ h1 -> X2[:,256:512]
    gemm_wmma<16, true, false, false, false><<<M0 / 128, 256, 0, stream>>>(
        (const void*)mask0, M1, h1T, M1,
        (void*)(X2 + DH), 2 * DH, ((_Float16*)nullptr), 0, M1);

    // stage 4: out = relu(X2 @ W2^T), f32 output
    gemm_wmma<16, false, true, true, false><<<M0 / 128, 256, 0, stream>>>(
        (const void*)X2, 2 * DH, W2h, 2 * DH,
        (void*)out, DOUT, ((_Float16*)nullptr), 0, 2 * DH);
}
